// TMSA_46789373722701
// MI455X (gfx1250) — matrix operand order repaired and checked by reading, not compiled, not run on hardware
//
#include <hip/hip_runtime.h>
#include <hip/hip_bf16.h>
#include <cmath>

// Problem constants (match reference)
#define D_MODEL 1024
#define NHEAD   16
#define DK      64
#define BATCH   4
#define SEQ     2048
#define M_TOTAL (BATCH*SEQ)   // 8192 rows in projection GEMM
#define KCAT    (2*D_MODEL)   // 2048 concatenated K (Xs|Xt)

// GEMM tiling: block 256x64, 8 waves each owning a 64x32 tile (8 WMMAs/chunk)
#define TM 256
#define TN 64
#define TK 32

typedef __attribute__((ext_vector_type(16))) __bf16 v16bf;
typedef __attribute__((ext_vector_type(8)))  float  v8f;

union FragBF {
  uint4 u[2];
  v16bf v;
};

__device__ __forceinline__ unsigned short f2bf(float x) {
  unsigned int u = __float_as_uint(x);
  u += 0x7FFFu + ((u >> 16) & 1u);   // round-to-nearest-even
  return (unsigned short)(u >> 16);
}

// ---------------------------------------------------------------------------
// CDNA5 async global->LDS copy (ASYNCcnt-tracked), with synchronous fallback.
// Builtin prototype (from hipcc diagnostic): (AS1 v4i*, AS3 v4i*, Imm, Imm).
// ---------------------------------------------------------------------------
#if defined(__has_builtin)
#if __has_builtin(__builtin_amdgcn_global_load_async_to_lds_b128)
#define HAVE_ASYNC_LDS 1
#endif
#endif
#ifndef HAVE_ASYNC_LDS
#define HAVE_ASYNC_LDS 0
#endif

typedef int v4i_vs __attribute__((vector_size(16)));

__device__ __forceinline__ void async_copy_b128(const void* g, void* l) {
#if HAVE_ASYNC_LDS
  __builtin_amdgcn_global_load_async_to_lds_b128(
      (__attribute__((address_space(1))) v4i_vs*)(g),
      (__attribute__((address_space(3))) v4i_vs*)(l), 0, 0);
#else
  *(uint4*)l = *(const uint4*)g;
#endif
}

__device__ __forceinline__ void async_wait0() {
#if HAVE_ASYNC_LDS
#if defined(__has_builtin) && __has_builtin(__builtin_amdgcn_s_wait_asynccnt)
  __builtin_amdgcn_s_wait_asynccnt(0);
#else
  asm volatile("s_wait_asynccnt 0x0" ::: "memory");
#endif
#endif
}

// ---------------------------------------------------------------------------
// Stage 1: f32 -> bf16 conversion, concatenating "s" and "t" halves along K.
// out is (rows, 2*cols) bf16 row-major.
// ---------------------------------------------------------------------------
__global__ void concat_to_bf16(const float* __restrict__ A,
                               const float* __restrict__ B,
                               unsigned short* __restrict__ out,
                               int rows, int cols) {
  int total = rows * 2 * cols;
  for (int i = blockIdx.x * blockDim.x + threadIdx.x; i < total;
       i += gridDim.x * blockDim.x) {
    int r = i / (2 * cols);
    int c = i - r * (2 * cols);
    float v = (c < cols) ? A[r * cols + c] : B[r * cols + (c - cols)];
    out[i] = f2bf(v);
  }
}

__global__ void bias_sum_kernel(const float* __restrict__ a,
                                const float* __restrict__ b,
                                float* __restrict__ out) {
  int i = blockIdx.x * blockDim.x + threadIdx.x;
  if (i < D_MODEL) out[i] = a[i] + b[i];
}

// ---------------------------------------------------------------------------
// Stage 2: projection GEMM.  D = Xcat(8192x2048) * Wcat^T + bias (per proj).
// blockIdx.z selects Q/K/V.  Output bf16:
//   Q,K -> (b,h,s,d) row-major;  V -> (b,h,d,s) (pre-transposed for PV WMMA).
// Double-buffered LDS, async global->LDS prefetch when available.
// ---------------------------------------------------------------------------
__device__ __forceinline__ void issue_tiles(const unsigned short* __restrict__ X,
                                            const unsigned short* __restrict__ W,
                                            unsigned short* at,
                                            unsigned short* bt,
                                            int tid, int m0, int n0, int kk) {
  // A tile: 256x32 halfs = 1024 uint4 ; 4 per thread
#pragma unroll
  for (int it = 0; it < 4; ++it) {
    int id  = tid + it * 256;
    int row = id >> 2;
    int col = (id & 3) * 8;
    async_copy_b128(&X[(size_t)(m0 + row) * KCAT + kk + col],
                    &at[row * TK + col]);
  }
  // B tile: 64x32 halfs = 256 uint4 ; 1 per thread
  {
    int row = tid >> 2;
    int col = (tid & 3) * 8;
    async_copy_b128(&W[(size_t)(n0 + row) * KCAT + kk + col],
                    &bt[row * TK + col]);
  }
}

__global__ __launch_bounds__(256)
void proj_gemm(const unsigned short* __restrict__ X,
               const unsigned short* __restrict__ Wq,
               const unsigned short* __restrict__ Wk,
               const unsigned short* __restrict__ Wv,
               const float* __restrict__ bsum,
               unsigned short* __restrict__ Qb,
               unsigned short* __restrict__ Kb,
               unsigned short* __restrict__ Vb) {
  __shared__ __align__(16) unsigned short At[2][TM * TK];  // 2 x 16 KB
  __shared__ __align__(16) unsigned short Bt[2][TN * TK];  // 2 x  4 KB

  const int proj = blockIdx.z;
  const unsigned short* W = (proj == 0) ? Wq : (proj == 1) ? Wk : Wv;
  const float* bias = bsum + proj * D_MODEL;

  const int tid  = threadIdx.x;
  const int lane = tid & 31;
  const int wave = tid >> 5;
  const int wm   = wave & 3;   // M sub-tile 0..3 (64 rows each)
  const int wn   = wave >> 2;  // N sub-tile 0..1 (32 cols each)
  const int hf   = lane >> 4;  // lane half (wave32)
  const int l15  = lane & 15;

  const int m0 = blockIdx.x * TM;
  const int n0 = blockIdx.y * TN;

  v8f acc[4][2];
#pragma unroll
  for (int a = 0; a < 4; ++a)
#pragma unroll
    for (int c = 0; c < 2; ++c)
#pragma unroll
      for (int i = 0; i < 8; ++i) acc[a][c][i] = 0.f;

  const int nch = KCAT / TK;   // 64 K-chunks
  issue_tiles(X, W, At[0], Bt[0], tid, m0, n0, 0);

  for (int kk = 0; kk < nch; ++kk) {
    const int cur = kk & 1;
    // (a) my prefetch into buf[cur] landed; (b) everyone is done reading
    // buf[cur^1] from last iteration -> safe to overwrite it next.
    async_wait0();
    __syncthreads();
    if (kk + 1 < nch)
      issue_tiles(X, W, At[cur ^ 1], Bt[cur ^ 1], tid, m0, n0, (kk + 1) * TK);

    const unsigned short* at = At[cur];
    const unsigned short* bt = Bt[cur];

    // Fragments per ISA 16-bit layouts (row-major along K in LDS).
    FragBF af[4], bfr[2];
#pragma unroll
    for (int mi = 0; mi < 4; ++mi) {
      int r  = wm * 64 + mi * 16 + l15;     // A: lane = M row
      int ko = hf * 8;                       // K 0-7/16-23 vs 8-15/24-31
      af[mi].u[0] = *(const uint4*)&at[r * TK + ko];
      af[mi].u[1] = *(const uint4*)&at[r * TK + 16 + ko];
    }
#pragma unroll
    for (int ni = 0; ni < 2; ++ni) {
      int r  = wn * 32 + ni * 16 + l15;     // B: lane = N col
      int kb = hf * 8;                       // K 0-7/16-23 vs 8-15/24-31
      bfr[ni].u[0] = *(const uint4*)&bt[r * TK + kb];
      bfr[ni].u[1] = *(const uint4*)&bt[r * TK + 16 + kb];
    }
#pragma unroll
    for (int mi = 0; mi < 4; ++mi)
#pragma unroll
      for (int ni = 0; ni < 2; ++ni)
        acc[mi][ni] = __builtin_amdgcn_wmma_f32_16x16x32_bf16(
            false, af[mi].v, false, bfr[ni].v, (short)0, acc[mi][ni],
            false, false);
  }

  // Epilogue: bias add, bf16 convert, layout-aware scatter.
#pragma unroll
  for (int mi = 0; mi < 4; ++mi) {
#pragma unroll
    for (int ni = 0; ni < 2; ++ni) {
      int n  = n0 + wn * 32 + ni * 16 + l15;   // D/C layout: lane = N
      float bv = bias[n];
      int h = n >> 6;
      int d = n & 63;
#pragma unroll
      for (int r = 0; r < 8; ++r) {            // VGPR r -> M row r(+8*half)
        int m = m0 + wm * 64 + mi * 16 + r + 8 * hf;
        int bb = m >> 11;
        int s  = m & 2047;
        int bh = bb * NHEAD + h;
        unsigned short val = f2bf(acc[mi][ni][r] + bv);
        if (proj == 0)      Qb[((size_t)bh * SEQ + s) * DK + d] = val;
        else if (proj == 1) Kb[((size_t)bh * SEQ + s) * DK + d] = val;
        else                Vb[((size_t)bh * DK + d) * SEQ + s] = val;  // V^T
      }
    }
  }
}

// ---------------------------------------------------------------------------
// Stage 3: flash attention.  One wave owns one 16-query tile of one (b,h).
// 32 keys per iteration: 4 QK^T WMMAs, half-wave shuffle softmax,
// P routed through a 1KB/wave LDS tile for C->A relayout, 4 PV WMMAs.
// ---------------------------------------------------------------------------
__global__ __launch_bounds__(256)
void attn_kernel(const unsigned short* __restrict__ Qb,
                 const unsigned short* __restrict__ Kb,
                 const unsigned short* __restrict__ Vb,
                 const float* __restrict__ attn_bias,
                 float* __restrict__ out) {
  __shared__ __align__(16) unsigned short Pt[8][16 * 32];

  const int lane = threadIdx.x & 31;
  const int wave = threadIdx.x >> 5;
  const int tile = blockIdx.x * 8 + wave;   // 8192 q-tiles total
  const int bh = tile >> 7;                 // 128 q-tiles per (b,h)
  const int qt = tile & 127;
  const int bb = bh >> 4;
  const int h  = bh & 15;
  const int hf  = lane >> 4;
  const int l15 = lane & 15;
  const float abias = attn_bias[0];

  unsigned short* ptile = Pt[wave];

  // Q A-fragments: rows = queries, K split into two d-chunks of 32.
  FragBF qf[2];
  {
    int s = qt * 16 + l15;
    const unsigned short* qrow = Qb + ((size_t)bh * SEQ + s) * DK;
    int ko = hf * 8;
#pragma unroll
    for (int c = 0; c < 2; ++c) {
      qf[c].u[0] = *(const uint4*)&qrow[c * 32 + ko];
      qf[c].u[1] = *(const uint4*)&qrow[c * 32 + 16 + ko];
    }
  }

  v8f O[4];
  float mrow[8], lrow[8];
#pragma unroll
  for (int t = 0; t < 4; ++t)
#pragma unroll
    for (int i = 0; i < 8; ++i) O[t][i] = 0.f;
#pragma unroll
  for (int r = 0; r < 8; ++r) { mrow[r] = -INFINITY; lrow[r] = 0.f; }

  for (int kc = 0; kc < SEQ / 32; ++kc) {
    v8f s0, s1;
#pragma unroll
    for (int i = 0; i < 8; ++i) { s0[i] = 0.f; s1[i] = 0.f; }

    // Scores for 32 keys (two 16-key groups), K over d in chunks of 32.
    {
      int key0 = kc * 32 + l15;            // B-frag: lane = key (N)
      const unsigned short* k0 = Kb + ((size_t)bh * SEQ + key0) * DK;
      const unsigned short* k1 = Kb + ((size_t)bh * SEQ + key0 + 16) * DK;
      int kb = hf * 8;                     // B-frag: half-wave splits K
#pragma unroll
      for (int c = 0; c < 2; ++c) {
        FragBF f0, f1;
        f0.u[0] = *(const uint4*)&k0[c * 32 + kb];
        f0.u[1] = *(const uint4*)&k0[c * 32 + 16 + kb];
        f1.u[0] = *(const uint4*)&k1[c * 32 + kb];
        f1.u[1] = *(const uint4*)&k1[c * 32 + 16 + kb];
        s0 = __builtin_amdgcn_wmma_f32_16x16x32_bf16(false, qf[c].v, false, f0.v,
                                                     (short)0, s0, false, false);
        s1 = __builtin_amdgcn_wmma_f32_16x16x32_bf16(false, qf[c].v, false, f1.v,
                                                     (short)0, s1, false, false);
      }
    }

    // Online softmax. C layout: VGPR r = row (r + 8*half), lanes = 16 keys,
    // so row reductions are xor-butterflies over masks 1,2,4,8 (stay in half).
    float alpha[8];
#pragma unroll
    for (int r = 0; r < 8; ++r) {
      float a0 = s0[r] * 0.125f + abias;    // 1/sqrt(64)
      float a1 = s1[r] * 0.125f + abias;
      float t = fmaxf(a0, a1);
#pragma unroll
      for (int msk = 1; msk < 16; msk <<= 1)
        t = fmaxf(t, __shfl_xor(t, msk, 32));
      float mn = fmaxf(mrow[r], t);
      float p0 = __expf(a0 - mn);
      float p1 = __expf(a1 - mn);
      float rs = p0 + p1;
#pragma unroll
      for (int msk = 1; msk < 16; msk <<= 1)
        rs += __shfl_xor(rs, msk, 32);
      alpha[r] = __expf(mrow[r] - mn);
      lrow[r] = lrow[r] * alpha[r] + rs;
      mrow[r] = mn;
      int row = r + 8 * hf;
      ptile[row * 32 + l15]      = f2bf(p0);   // C->A relayout via LDS
      ptile[row * 32 + 16 + l15] = f2bf(p1);   // (DS ops in-order per wave)
    }
#pragma unroll
    for (int t = 0; t < 4; ++t)
#pragma unroll
      for (int r = 0; r < 8; ++r) O[t][r] *= alpha[r];

    // P as A-matrix (16 queries x 32 keys).
    FragBF pf;
    {
      int ko = hf * 8;
      pf.u[0] = *(const uint4*)&ptile[l15 * 32 + ko];
      pf.u[1] = *(const uint4*)&ptile[l15 * 32 + 16 + ko];
    }

    // PV: B-frags straight from pre-transposed V (d-major, keys contiguous).
#pragma unroll
    for (int t = 0; t < 4; ++t) {
      FragBF vf;
      int n = t * 16 + l15;                 // lane = output dim (N)
      const unsigned short* vrow =
          Vb + ((size_t)bh * DK + n) * SEQ + kc * 32 + hf * 8;
      vf.u[0] = *(const uint4*)&vrow[0];
      vf.u[1] = *(const uint4*)&vrow[16];
      O[t] = __builtin_amdgcn_wmma_f32_16x16x32_bf16(false, pf.v, false, vf.v,
                                                     (short)0, O[t], false, false);
    }
  }

  // Normalize and write (B,S,H*dk) f32.
#pragma unroll
  for (int t = 0; t < 4; ++t) {
    int d = t * 16 + l15;
#pragma unroll
    for (int r = 0; r < 8; ++r) {
      int s = qt * 16 + r + 8 * hf;
      out[((size_t)bb * SEQ + s) * D_MODEL + h * DK + d] = O[t][r] / lrow[r];
    }
  }
}

// ---------------------------------------------------------------------------
// Host launcher.  Workspace layout (bf16 unless noted), ~92 MB total:
//   Xcat  8192x2048            (32 MB)
//   Wq,Wk,Wv  1024x2048 each   (12 MB)
//   bsum  3x1024 f32           (12 KB)
//   Qb,Kb (b,h,s,d), Vb (b,h,d,s)  16 MB each
// ---------------------------------------------------------------------------
extern "C" void kernel_launch(void* const* d_in, const int* in_sizes, int n_in,
                              void* d_out, int out_size, void* d_ws, size_t ws_size,
                              hipStream_t stream) {
  const float* Xs   = (const float*)d_in[0];
  const float* Xt   = (const float*)d_in[1];
  const float* W_qs = (const float*)d_in[2];
  const float* b_qs = (const float*)d_in[3];
  const float* W_qt = (const float*)d_in[4];
  const float* b_qt = (const float*)d_in[5];
  const float* W_ks = (const float*)d_in[6];
  const float* b_ks = (const float*)d_in[7];
  const float* W_kt = (const float*)d_in[8];
  const float* b_kt = (const float*)d_in[9];
  const float* W_vs = (const float*)d_in[10];
  const float* b_vs = (const float*)d_in[11];
  const float* W_vt = (const float*)d_in[12];
  const float* b_vt = (const float*)d_in[13];
  const float* attn_bias = (const float*)d_in[14];

  unsigned short* Xcat = (unsigned short*)d_ws;
  unsigned short* Wq = Xcat + (size_t)M_TOTAL * KCAT;
  unsigned short* Wk = Wq + (size_t)D_MODEL * KCAT;
  unsigned short* Wv = Wk + (size_t)D_MODEL * KCAT;
  float* bsum = (float*)(Wv + (size_t)D_MODEL * KCAT);
  unsigned short* Qb = (unsigned short*)(bsum + 3 * D_MODEL);
  unsigned short* Kb = Qb + (size_t)M_TOTAL * D_MODEL;
  unsigned short* Vb = Kb + (size_t)M_TOTAL * D_MODEL;

  // Stage 1: bf16 conversion + concat + bias sums
  concat_to_bf16<<<8192, 256, 0, stream>>>(Xs, Xt, Xcat, M_TOTAL, D_MODEL);
  concat_to_bf16<<<2048, 256, 0, stream>>>(W_qs, W_qt, Wq, D_MODEL, D_MODEL);
  concat_to_bf16<<<2048, 256, 0, stream>>>(W_ks, W_kt, Wk, D_MODEL, D_MODEL);
  concat_to_bf16<<<2048, 256, 0, stream>>>(W_vs, W_vt, Wv, D_MODEL, D_MODEL);
  bias_sum_kernel<<<4, 256, 0, stream>>>(b_qs, b_qt, bsum);
  bias_sum_kernel<<<4, 256, 0, stream>>>(b_ks, b_kt, bsum + D_MODEL);
  bias_sum_kernel<<<4, 256, 0, stream>>>(b_vs, b_vt, bsum + 2 * D_MODEL);

  // Stage 2: Q/K/V projection GEMMs (z selects projection)
  dim3 ggrid(M_TOTAL / TM, D_MODEL / TN, 3);
  proj_gemm<<<ggrid, 256, 0, stream>>>(Xcat, Wq, Wk, Wv, bsum, Qb, Kb, Vb);

  // Stage 3: attention (one wave per 16-query tile; 8 waves/block)
  int qtiles = BATCH * NHEAD * (SEQ / 16);   // 8192
  attn_kernel<<<qtiles / 8, 256, 0, stream>>>(Qb, Kb, Vb, attn_bias,
                                              (float*)d_out);
}
